// Model_59253368816326
// MI455X (gfx1250) — matrix operand order repaired and checked by reading, not compiled, not run on hardware
//
#include <hip/hip_runtime.h>

typedef _Float16 half_t;
typedef _Float16 v16h __attribute__((ext_vector_type(16)));
typedef _Float16 v8h  __attribute__((ext_vector_type(8)));
typedef float    v8f  __attribute__((ext_vector_type(8)));
typedef unsigned int u32x4 __attribute__((ext_vector_type(4)));
typedef int      i32x4 __attribute__((ext_vector_type(4)));
typedef int      i32x8 __attribute__((ext_vector_type(8)));

#define EMB    64
#define IND    192          // 3*EMB
#define TD     128
#define TD3    384
#define NCOL   512          // 384 (iou) + 128 (f)
#define NTREES 256
#define NODES  511
#define NTN    (NTREES * NODES)   // 130816 nodes per side
#define NT2    (2 * NTN)          // 261632 rows total (side a then side b)

#if __has_builtin(__builtin_amdgcn_tensor_load_to_lds) && \
    __has_builtin(__builtin_amdgcn_s_wait_tensorcnt)
#define USE_TDM 1
#endif

__device__ __forceinline__ float sigf(float x) { return 1.0f / (1.0f + __expf(-x)); }

__device__ __forceinline__ v8f wmma16(v16h a, v16h b, v8f c) {
  // D = A(16x32 f16) * B(32x16 f16) + C(16x16 f32)
  return __builtin_amdgcn_wmma_f32_16x16x32_f16(false, a, false, b, (short)0, c, false, false);
}

// Build an A fragment from a row-major f16 LDS row pointer already offset to
// (row, ktile*32 + (lane>>4)*8). Halves [0..7] = K..K+7, [8..15] = K+16..K+23.
__device__ __forceinline__ v16h lds_a_frag(const half_t* p) {
  const v8h* q = (const v8h*)p;
  v8h lo = q[0];     // 16B aligned ds_load
  v8h hi = q[2];     // +16 halves
  return __builtin_shufflevector(lo, hi, 0,1,2,3,4,5,6,7,8,9,10,11,12,13,14,15);
}

__device__ __forceinline__ v16h b_frag(const half_t* pack, int kt, int nt, int ntiles, int lane) {
  const v16h* bp = (const v16h*)(pack + ((((size_t)(kt * ntiles + nt)) * 32 + lane) << 4));
  return *bp;        // 32B: two global_load_b128
}

#ifdef USE_TDM
// One-shot 1D TDM copy global -> LDS. `bytes` must be a multiple of 8 and
// (bytes/8) < 65536. D# per CDNA5 ISA:
//  group0: count=1 | lds_addr[63:32] | global_addr[120:64] | type=2[127:126]
//  group1: data_size=3 (8B) [17:16], tensor_dim0[79:48], tensor_dim1[111:80]=1,
//          tile_dim0[127:112], tile_dim1[143:128]=1, tensor_dim0_stride[207:160]
__device__ __forceinline__ void tdm_load_1d(unsigned lds_off, const void* gptr,
                                            unsigned bytes) {
  unsigned long long ga = (unsigned long long)gptr;
  unsigned units = bytes >> 3;             // 8-byte units
  u32x4 g0;
  g0[0] = 1u;                              // count = 1, user descriptor
  g0[1] = lds_off;                         // LDS byte address
  g0[2] = (unsigned)ga;                    // global addr [31:0]
  g0[3] = (unsigned)((ga >> 32) & 0x01FFFFFFu) | (2u << 30);  // addr[56:32] | type=2
  i32x8 g1;
  g1[0] = (int)(3u << 16);                 // data_size = 8B; no multicast/pad/iterate
  g1[1] = (int)(units << 16);              // tensor_dim0[15:0] @ bits 63:48
  g1[2] = (int)(((units >> 16) & 0xFFFFu) | (1u << 16)); // dim0[31:16], tensor_dim1=1
  g1[3] = (int)(units << 16);              // tile_dim0 @ bits 127:112
  g1[4] = 1;                               // tile_dim1 = 1 (single row)
  g1[5] = (int)units;                      // tensor_dim0_stride low 32
  g1[6] = 0;
  g1[7] = 0;
  i32x4 z4 = {0, 0, 0, 0};
#if __clang_major__ >= 23
  i32x8 z8 = {0, 0, 0, 0, 0, 0, 0, 0};
  __builtin_amdgcn_tensor_load_to_lds(g0, g1, z4, z4, z8, 0);
#else
  __builtin_amdgcn_tensor_load_to_lds(g0, g1, z4, z4, 0);
#endif
}
#endif

// ---------------------------------------------------------------------------
// Kernel 0: pack W (=[W_iou | W_f], 192x512), U_iou (128x384), U_f (128x128)
// into f16 WMMA B-fragment order.
// element e -> i = e&15, lane = (e>>4)&31, frag = e>>9, nt = frag%ntiles,
// kt = frag/ntiles; col = nt*16 + (lane&15); k = kt*32 + (lane>>4)*8 + (i&7) + (i>>3)*16.
// ---------------------------------------------------------------------------
__global__ void pack_weights(const float* __restrict__ W_iou, const float* __restrict__ W_f,
                             const float* __restrict__ U_iou, const float* __restrict__ U_f,
                             half_t* __restrict__ Wp, half_t* __restrict__ Up,
                             half_t* __restrict__ Ufp) {
  int e = blockIdx.x * blockDim.x + threadIdx.x;   // 0 .. 163839
  const int NW  = IND / 32 * (NCOL / 16) * 512;    // 6*32*512 = 98304
  const int NU  = TD / 32 * (TD3 / 16) * 512;      // 4*24*512 = 49152
  half_t* dst; int ntiles; int idx; int mode;
  if (e < NW)            { dst = Wp;  ntiles = NCOL / 16; idx = e;            mode = 0; }
  else if (e < NW + NU)  { dst = Up;  ntiles = TD3  / 16; idx = e - NW;       mode = 1; }
  else                   { dst = Ufp; ntiles = TD   / 16; idx = e - NW - NU;  mode = 2; }
  int i    = idx & 15;
  int lane = (idx >> 4) & 31;
  int frag = idx >> 9;
  int nt = frag % ntiles;
  int kt = frag / ntiles;
  int col = nt * 16 + (lane & 15);
  int k   = kt * 32 + ((lane >> 4) << 3) + (i & 7) + ((i >> 3) << 4);
  float v;
  if (mode == 0)      v = (col < TD3) ? W_iou[k * TD3 + col] : W_f[k * TD + (col - TD3)];
  else if (mode == 1) v = U_iou[k * TD3 + col];
  else                v = U_f[k * TD + col];
  dst[idx] = (half_t)v;
}

// ---------------------------------------------------------------------------
// Kernel 1: feat gather + [NT2,192] @ [192,512] -> wiou16 [NT2,384], wf16 [NT2,128]
// 16 rows per workgroup, 8 waves. K-outer loop: each k-step's 32KB B-fragment
// panel is DMA'd into LDS once per workgroup via the Tensor Data Mover, then
// all 8 waves consume it (4 column tiles each, accumulators live across k).
// ---------------------------------------------------------------------------
__global__ void embed_gemm(const int* __restrict__ tok_a, const int* __restrict__ srt_a,
                           const float* __restrict__ cst_a,
                           const int* __restrict__ tok_b, const int* __restrict__ srt_b,
                           const float* __restrict__ cst_b,
                           const float* __restrict__ emb, const float* __restrict__ sort_tab,
                           const half_t* __restrict__ Wp,
                           const float* __restrict__ b_iou, const float* __restrict__ b_f,
                           half_t* __restrict__ wiou16, half_t* __restrict__ wf16) {
  __shared__ half_t As[16][200];    // 192 + pad, row stride 400B (16B aligned)
  __shared__ half_t Bp[32 * 512];   // 32KB: one k-step's B fragments (nt=0..31)
  const int tid = threadIdx.x;
  const int g0  = blockIdx.x * 16;

  // gather 16x192 feature tile (f32 -> f16)
  for (int idx = tid; idx < 16 * IND; idx += 256) {
    int r = idx / IND, cc = idx - r * IND;
    int g = g0 + r;
    int side = (g >= NTN);
    int node = g - side * NTN;
    const int*   tok = side ? tok_b : tok_a;
    const int*   srt = side ? srt_b : srt_a;
    const float* cst = side ? cst_b : cst_a;
    float v;
    if (cc < EMB)            v = emb[(size_t)tok[node] * EMB + cc];
    else if (cc < 2 * EMB)   v = sort_tab[(size_t)srt[node] * EMB + (cc - EMB)];
    else                     v = cst[(size_t)node * EMB + (cc - 2 * EMB)];
    As[r][cc] = (half_t)v;
  }

  const int wave = tid >> 5, lane = tid & 31;
  const int arow = lane & 15, ahi = lane >> 4;
  v8f acc[4] = {};

#ifdef USE_TDM
  const unsigned bp_off = (unsigned)(unsigned long long)(void*)&Bp[0];
#endif

  for (int kt = 0; kt < 6; ++kt) {
    __syncthreads();   // previous panel fully consumed (also covers As on kt==0)
#ifdef USE_TDM
    if (wave == 0) {
      tdm_load_1d(bp_off, Wp + (size_t)kt * 32 * 512, 32 * 512 * sizeof(half_t));
      __builtin_amdgcn_s_wait_tensorcnt(0);
    }
#else
    for (int idx = tid; idx < (32 * 512 * (int)sizeof(half_t)) / 16; idx += 256)
      ((uint4*)Bp)[idx] = ((const uint4*)(Wp + (size_t)kt * 32 * 512))[idx];
#endif
    __syncthreads();   // panel visible to all waves

    v16h a = lds_a_frag(&As[arow][kt * 32 + ahi * 8]);
#pragma unroll
    for (int q = 0; q < 4; ++q) {
      int nt = wave * 4 + q;               // 0..31
      v16h b = *(const v16h*)&Bp[((nt * 32) + lane) << 4];
      acc[q] = wmma16(a, b, acc[q]);
    }
  }

#pragma unroll
  for (int q = 0; q < 4; ++q) {
    int nt = wave * 4 + q;
    int col  = nt * 16 + (lane & 15);
    float bs = (col < TD3) ? b_iou[col] : b_f[col - TD3];
    int rbase = (lane >> 4) * 8;
#pragma unroll
    for (int j = 0; j < 8; ++j) {
      int g = g0 + rbase + j;
      float v = acc[q][j] + bs;
      if (col < TD3) wiou16[(size_t)g * TD3 + col] = (half_t)v;
      else           wf16[(size_t)g * TD + (col - TD3)] = (half_t)v;
    }
  }
}

// ---------------------------------------------------------------------------
// Kernel 2: leaves (level 0): c = sig(i)*tanh(u); h = sig(o)*tanh(c)
// ---------------------------------------------------------------------------
__global__ void leaf_update(const half_t* __restrict__ wiou16,
                            float* __restrict__ c32, half_t* __restrict__ h16) {
  int idx = blockIdx.x * blockDim.x + threadIdx.x;   // < 2*256*256*128 = 2^24
  int d    = idx & 127;
  int leaf = (idx >> 7) & 255;
  int tree = (idx >> 15) & 255;
  int side = idx >> 23;
  int g = side * NTN + tree * NODES + leaf;
  float i = (float)wiou16[(size_t)g * TD3 + d];
  float o = (float)wiou16[(size_t)g * TD3 + TD + d];
  float u = (float)wiou16[(size_t)g * TD3 + 2 * TD + d];
  float c = sigf(i) * tanhf(u);
  float h = sigf(o) * tanhf(c);
  c32[(size_t)g * TD + d] = c;
  h16[(size_t)g * TD + d] = (half_t)h;
}

// ---------------------------------------------------------------------------
// Kernel 3: one tree level. 16 parents (32 children) per workgroup, 8 waves.
// GEMM1: ch[32x128] @ U_f[128x128]     (16 tiles, 2 per wave)
// GEMM2: hsum[16x128] @ U_iou[128x384] (24 tiles, 3 per wave)
// then the elementwise TreeLSTM cell update.
// ---------------------------------------------------------------------------
__global__ void level_step(half_t* __restrict__ h16, float* __restrict__ c32,
                           const half_t* __restrict__ wiou16, const half_t* __restrict__ wf16,
                           const half_t* __restrict__ Ufp, const half_t* __restrict__ Up,
                           int l) {
  __shared__ half_t hc[32][TD];      // child h        8 KB
  __shared__ half_t hs[16][TD];      // h_sum          4 KB
  __shared__ float  fpre[32][TD];    // ch@U_f        16 KB
  __shared__ float  iou[16][TD3];    // hsum@U_iou    24 KB
  __shared__ int    pIdx[16];
  __shared__ int    cIdx[32];

  const int tid = threadIdx.x;
  const int L = 1 << (8 - l);
  const int offP = 512 - (1 << (9 - l));
  const int offC = 512 - (1 << (10 - l));

  if (tid < 16) {
    int pf   = blockIdx.x * 16 + tid;
    int side = pf >> (16 - l);
    int rem  = pf & ((1 << (16 - l)) - 1);
    int tree = rem >> (8 - l);
    int j    = rem & (L - 1);
    int base = side * NTN + tree * NODES;
    pIdx[tid]         = base + offP + j;
    cIdx[2 * tid]     = base + offC + 2 * j;
    cIdx[2 * tid + 1] = base + offC + 2 * j + 1;
  }
  __syncthreads();

  { // stage 32 child-h rows: each thread copies one 32B segment
    int e = tid >> 3, seg = tid & 7;
    const uint4* src = (const uint4*)(h16 + (size_t)cIdx[e] * TD + seg * 16);
    uint4* dst = (uint4*)&hc[e][seg * 16];
    dst[0] = src[0];
    dst[1] = src[1];
  }
  __syncthreads();

  for (int idx = tid; idx < 16 * TD; idx += 256) {
    int r = idx >> 7, d = idx & 127;
    hs[r][d] = (half_t)((float)hc[2 * r][d] + (float)hc[2 * r + 1][d]);
  }
  __syncthreads();

  const int wave = tid >> 5, lane = tid & 31;
  const int arow = lane & 15, ahi = lane >> 4;
  const int crow = (lane >> 4) * 8, ccol = lane & 15;

  // GEMM1: fpre = hc @ U_f
  for (int t = 0; t < 2; ++t) {
    int tileId = wave * 2 + t;            // 0..15
    int mt = tileId >> 3, nt = tileId & 7;
    v8f acc = {};
#pragma unroll
    for (int kt = 0; kt < 4; ++kt) {
      v16h a = lds_a_frag(&hc[mt * 16 + arow][kt * 32 + ahi * 8]);
      v16h b = b_frag(Ufp, kt, nt, 8, lane);
      acc = wmma16(a, b, acc);
    }
#pragma unroll
    for (int j = 0; j < 8; ++j)
      fpre[mt * 16 + crow + j][nt * 16 + ccol] = acc[j];
  }

  // GEMM2: iou = hs @ U_iou
  for (int t = 0; t < 3; ++t) {
    int nt = wave * 3 + t;                // 0..23
    v8f acc = {};
#pragma unroll
    for (int kt = 0; kt < 4; ++kt) {
      v16h a = lds_a_frag(&hs[arow][kt * 32 + ahi * 8]);
      v16h b = b_frag(Up, kt, nt, 24, lane);
      acc = wmma16(a, b, acc);
    }
#pragma unroll
    for (int j = 0; j < 8; ++j)
      iou[crow + j][nt * 16 + ccol] = acc[j];
  }
  __syncthreads();

  // elementwise cell update
  for (int idx = tid; idx < 16 * TD; idx += 256) {
    int r = idx >> 7, d = idx & 127;
    int p = pIdx[r], c0 = cIdx[2 * r], c1 = cIdx[2 * r + 1];
    float wf = (float)wf16[(size_t)p * TD + d];
    float f0 = sigf(wf + fpre[2 * r][d]);
    float f1 = sigf(wf + fpre[2 * r + 1][d]);
    float csum = f0 * c32[(size_t)c0 * TD + d] + f1 * c32[(size_t)c1 * TD + d];
    float gi = (float)wiou16[(size_t)p * TD3 + d]          + iou[r][d];
    float go = (float)wiou16[(size_t)p * TD3 + TD + d]     + iou[r][TD + d];
    float gu = (float)wiou16[(size_t)p * TD3 + 2 * TD + d] + iou[r][2 * TD + d];
    float cn = sigf(gi) * tanhf(gu) + csum;
    float hn = sigf(go) * tanhf(cn);
    c32[(size_t)p * TD + d] = cn;
    h16[(size_t)p * TD + d] = (half_t)hn;
  }
}

// ---------------------------------------------------------------------------
// Kernel 4: classifier head (tiny; plain VALU). One block per tree, 128 thr.
// ---------------------------------------------------------------------------
__global__ void classifier(const half_t* __restrict__ h16,
                           const float* __restrict__ fc1_w, const float* __restrict__ fc1_b,
                           const float* __restrict__ fc2_w, const float* __restrict__ fc2_b,
                           float* __restrict__ out) {
  __shared__ float hcat[257];
  __shared__ float red[128];
  __shared__ float hid[TD];
  const int b = blockIdx.x, tid = threadIdx.x;
  int ga = b * NODES + (NODES - 1);
  int gb = NTN + b * NODES + (NODES - 1);
  hcat[tid]       = (float)h16[(size_t)ga * TD + tid];
  hcat[128 + tid] = (float)h16[(size_t)gb * TD + tid];
  __syncthreads();
  red[tid] = hcat[tid] * hcat[128 + tid];
  __syncthreads();
  for (int s = 64; s > 0; s >>= 1) {
    if (tid < s) red[tid] += red[tid + s];
    __syncthreads();
  }
  if (tid == 0) hcat[256] = red[0];
  __syncthreads();
  float s = fc1_b[tid];
  for (int k = 0; k < 257; ++k) s += hcat[k] * fc1_w[k * TD + tid];
  hid[tid] = fmaxf(s, 0.0f);
  __syncthreads();
  if (tid < 2) {
    float lg = fc2_b[tid];
    for (int j = 0; j < TD; ++j) lg += hid[j] * fc2_w[j * 2 + tid];
    out[b * 2 + tid] = lg;
  }
}

// ---------------------------------------------------------------------------
extern "C" void kernel_launch(void* const* d_in, const int* in_sizes, int n_in,
                              void* d_out, int out_size, void* d_ws, size_t ws_size,
                              hipStream_t stream) {
  const int*   tok_a  = (const int*)d_in[0];
  const int*   srt_a  = (const int*)d_in[1];
  const float* cst_a  = (const float*)d_in[2];
  const int*   tok_b  = (const int*)d_in[3];
  const int*   srt_b  = (const int*)d_in[4];
  const float* cst_b  = (const float*)d_in[5];
  // d_in[6..11]: topology arrays + scalars (topology is analytic; unused)
  const float* emb      = (const float*)d_in[12];
  const float* sort_tab = (const float*)d_in[13];
  const float* W_iou = (const float*)d_in[14];
  const float* b_iou = (const float*)d_in[15];
  const float* U_iou = (const float*)d_in[16];
  const float* W_f   = (const float*)d_in[17];
  const float* b_f   = (const float*)d_in[18];
  const float* U_f   = (const float*)d_in[19];
  const float* fc1_w = (const float*)d_in[20];
  const float* fc1_b = (const float*)d_in[21];
  const float* fc2_w = (const float*)d_in[22];
  const float* fc2_b = (const float*)d_in[23];
  float* out = (float*)d_out;

  // workspace carve (256B aligned)
  char* w = (char*)d_ws;
  auto take = [&](size_t n) { char* p = w; w += (n + 255) & ~(size_t)255; return p; };
  half_t* wiou16 = (half_t*)take((size_t)NT2 * TD3 * sizeof(half_t));
  half_t* wf16   = (half_t*)take((size_t)NT2 * TD  * sizeof(half_t));
  half_t* h16    = (half_t*)take((size_t)NT2 * TD  * sizeof(half_t));
  float*  c32    = (float*) take((size_t)NT2 * TD  * sizeof(float));
  half_t* Wp     = (half_t*)take((size_t)IND * NCOL * sizeof(half_t));
  half_t* Up     = (half_t*)take((size_t)TD * TD3 * sizeof(half_t));
  half_t* Ufp    = (half_t*)take((size_t)TD * TD * sizeof(half_t));
  (void)in_sizes; (void)n_in; (void)out_size; (void)ws_size;

  pack_weights<<<640, 256, 0, stream>>>(W_iou, W_f, U_iou, U_f, Wp, Up, Ufp);

  embed_gemm<<<NT2 / 16, 256, 0, stream>>>(tok_a, srt_a, cst_a, tok_b, srt_b, cst_b,
                                           emb, sort_tab, Wp, b_iou, b_f, wiou16, wf16);

  leaf_update<<<(2 * NTREES * 256 * TD) / 256, 256, 0, stream>>>(wiou16, c32, h16);

  for (int l = 1; l <= 8; ++l) {
    int tiles = 32 << (8 - l);   // (2*256*2^(8-l)) / 16
    level_step<<<tiles, 256, 0, stream>>>(h16, c32, wiou16, wf16, Ufp, Up, l);
  }

  classifier<<<NTREES, 128, 0, stream>>>(h16, fc1_w, fc1_b, fc2_w, fc2_b, out);
}
